// REMSA_33629593928116
// MI455X (gfx1250) — matrix operand order repaired and checked by reading, not compiled, not run on hardware
//
#include <hip/hip_runtime.h>
#include <hip/hip_bf16.h>
#include <stdint.h>

typedef _Float16 v16h __attribute__((ext_vector_type(16)));
typedef float    v8f  __attribute__((ext_vector_type(8)));

#define N_TOK 64
#define QS 1032            // qkvp LDS row stride (f16 elems)
#define XS 264             // x / out-tile LDS row stride
#define AS 72              // attn-scratch / dwout row stride

// workspace byte offsets
#define OFF_WQ    0u        // packed f16 qkvp_w : u32[128][1024]
#define OFF_WPROJ 524288u   // packed f16 proj_w : u32[128][256]
#define OFF_PW    655360u   // f16 pw_w [256][32]
#define OFF_BIAS  671744u   // f32 bias [8][64][64]

#define SMEM_QKVP 33792
#define SMEM_SCR  165888
#define SMEM_BYTES 239616

union ABu { v16h v; uint32_t u[8]; _Float16 h[16]; };
union P2  { uint32_t u; _Float16 h[2]; };

static __device__ __forceinline__ v8f wmma16(const ABu& a, const ABu& b, v8f c) {
  return __builtin_amdgcn_wmma_f32_16x16x32_f16(false, a.v, false, b.v, (short)0, c, false, false);
}

// ---------------- pre-pass: weight conversion / bias gather ----------------
__global__ void remsa_prep(const float* __restrict__ qkvp_w,
                           const float* __restrict__ proj_w,
                           const float* __restrict__ pw_w,
                           const float* __restrict__ rpb,
                           const int*   __restrict__ rel_idx,
                           unsigned char* __restrict__ ws) {
  int id = blockIdx.x * 256 + threadIdx.x;
  uint32_t*  Wq  = (uint32_t*)(ws + OFF_WQ);
  uint32_t*  Wp  = (uint32_t*)(ws + OFF_WPROJ);
  _Float16*  pwh = (_Float16*)(ws + OFF_PW);
  float*     Bs  = (float*)(ws + OFF_BIAS);
  if (id < 131072) {                       // qkvp_w (256x1024) -> packed pairs
    int kp = id >> 10, j = id & 1023;
    P2 t; t.h[0] = (_Float16)qkvp_w[(2*kp)*1024 + j];
          t.h[1] = (_Float16)qkvp_w[(2*kp+1)*1024 + j];
    Wq[id] = t.u;
  } else if (id < 131072 + 32768) {        // proj_w (256x256) -> packed pairs
    int i2 = id - 131072;
    int kp = i2 >> 8, j = i2 & 255;
    P2 t; t.h[0] = (_Float16)proj_w[(2*kp)*256 + j];
          t.h[1] = (_Float16)proj_w[(2*kp+1)*256 + j];
    Wp[i2] = t.u;
  } else if (id < 131072 + 32768 + 8192) { // pw_w (256x32) -> f16
    int i3 = id - 131072 - 32768;
    pwh[i3] = (_Float16)pw_w[i3];
  } else if (id < 131072 + 32768 + 8192 + 32768) { // bias[h][q][k]
    int i4 = id - 131072 - 32768 - 8192;
    int h = i4 >> 12, q = (i4 >> 6) & 63, k = i4 & 63;
    Bs[i4] = rpb[rel_idx[q*64 + k]*8 + h];
  }
}

// ---------------- main fused kernel: one workgroup per window ----------------
__global__ __launch_bounds__(256)
void remsa_main(const float* __restrict__ x,
                const float* __restrict__ qkvp_b,
                const float* __restrict__ dw_w,
                const float* __restrict__ dw_b,
                const float* __restrict__ pw_b,
                const float* __restrict__ proj_b,
                const unsigned char* __restrict__ ws,
                float* __restrict__ out) {
  extern __shared__ __align__(16) unsigned char smem[];
  _Float16* xT  = (_Float16*)(smem);             // [64][XS] x tile, later out tile
  _Float16* qk  = (_Float16*)(smem + SMEM_QKVP); // [64][QS] qkvp f16
  _Float16* scr = (_Float16*)(smem + SMEM_SCR);  // per-wave attn A [64][AS]; later dwout [256][AS]

  const int tid  = threadIdx.x;
  const int wave = tid >> 5;
  const int lane = tid & 31;
  const int l16  = lane & 15;
  const int half = (lane >> 4) & 1;
  const size_t blk = blockIdx.x;

  const uint32_t* Wq    = (const uint32_t*)(ws + OFF_WQ);
  const uint32_t* Wp    = (const uint32_t*)(ws + OFF_WPROJ);
  const _Float16* pwW   = (const _Float16*)(ws + OFF_PW);
  const float*    BiasT = (const float*)(ws + OFF_BIAS);

  // ---- phase 1: load x (64x256 f32) -> f16 LDS ----
  {
    const float4* xg = (const float4*)(x + blk * (size_t)(N_TOK * 256));
    #pragma unroll
    for (int i = 0; i < 16; ++i) {
      int e = tid + i * 256;            // float4 index
      float4 v = xg[e];
      int n = e >> 6;
      int c = (e & 63) << 2;
      P2 t0, t1;
      t0.h[0] = (_Float16)v.x; t0.h[1] = (_Float16)v.y;
      t1.h[0] = (_Float16)v.z; t1.h[1] = (_Float16)v.w;
      *(uint32_t*)&xT[n * XS + c]     = t0.u;
      *(uint32_t*)&xT[n * XS + c + 2] = t1.u;
    }
  }
  __syncthreads();

  // ---- phase 2: qkvp GEMM (64x1024 = X(64x256) @ W(256x1024) + b), f16 -> LDS ----
  {
    const int colbase = wave * 128;
    #pragma unroll 1
    for (int mt = 0; mt < 4; ++mt) {
      ABu ax[8];
      #pragma unroll
      for (int ks = 0; ks < 8; ++ks)
        #pragma unroll
        for (int v = 0; v < 8; ++v) {
          int kb = ks*32 + (half<<4) + (v<<1);
          ax[ks].u[v] = *(const uint32_t*)&xT[(mt*16 + l16)*XS + kb];
        }
      #pragma unroll 1
      for (int jt = 0; jt < 8; ++jt) {
        int col = colbase + jt*16 + l16;
        float bb = qkvp_b[col];
        v8f acc;
        #pragma unroll
        for (int r = 0; r < 8; ++r) acc[r] = bb;
        #pragma unroll
        for (int ks = 0; ks < 8; ++ks) {
          ABu bw;
          #pragma unroll
          for (int v = 0; v < 8; ++v) {
            int kp = ks*16 + (half<<3) + v;
            bw.u[v] = Wq[kp*1024 + col];
          }
          acc = wmma16(ax[ks], bw, acc);
        }
        #pragma unroll
        for (int r = 0; r < 8; ++r)
          qk[(mt*16 + (half<<3) + r)*QS + col] = (_Float16)acc[r];
      }
    }
  }
  __syncthreads();

  // ---- phase 3: attention, wave = head ----
  {
    const int h = wave;
    _Float16* Asc = scr + wave * (64 * AS);
    ABu bk[4];                               // B = k^T tiles (K-dim = hd = 32)
    #pragma unroll
    for (int nt = 0; nt < 4; ++nt)
      #pragma unroll
      for (int v = 0; v < 8; ++v) {
        int kb = (half<<4) + (v<<1);
        bk[nt].u[v] = *(const uint32_t*)&qk[(nt*16 + l16)*QS + 256 + h*32 + kb];
      }
    #pragma unroll 1
    for (int mt = 0; mt < 4; ++mt) {
      ABu aq;
      #pragma unroll
      for (int v = 0; v < 8; ++v) {
        int kb = (half<<4) + (v<<1);
        aq.u[v] = *(const uint32_t*)&qk[(mt*16 + l16)*QS + h*32 + kb];
      }
      float S[4][8];
      #pragma unroll
      for (int nt = 0; nt < 4; ++nt) {
        v8f cc = {0.f,0.f,0.f,0.f,0.f,0.f,0.f,0.f};
        cc = wmma16(aq, bk[nt], cc);
        #pragma unroll
        for (int r = 0; r < 8; ++r) {
          int q = mt*16 + (half<<3) + r;
          S[nt][r] = cc[r] * 0.17677669529663689f + BiasT[h*4096 + q*64 + nt*16 + l16];
        }
      }
      #pragma unroll
      for (int r = 0; r < 8; ++r) {          // softmax across 64 keys (4 regs x 16 lanes)
        float m = fmaxf(fmaxf(S[0][r], S[1][r]), fmaxf(S[2][r], S[3][r]));
        m = fmaxf(m, __shfl_xor(m, 1, 32));
        m = fmaxf(m, __shfl_xor(m, 2, 32));
        m = fmaxf(m, __shfl_xor(m, 4, 32));
        m = fmaxf(m, __shfl_xor(m, 8, 32));
        float s0 = 0.f;
        #pragma unroll
        for (int nt = 0; nt < 4; ++nt) { S[nt][r] = __expf(S[nt][r] - m); s0 += S[nt][r]; }
        s0 += __shfl_xor(s0, 1, 32);
        s0 += __shfl_xor(s0, 2, 32);
        s0 += __shfl_xor(s0, 4, 32);
        s0 += __shfl_xor(s0, 8, 32);
        float inv = 1.0f / s0;
        int q = mt*16 + (half<<3) + r;
        #pragma unroll
        for (int nt = 0; nt < 4; ++nt)
          Asc[q*AS + nt*16 + l16] = (_Float16)(S[nt][r] * inv);
      }
    }
    // O = A(64x64) @ V(64x32)
    ABu bv[2][2];
    #pragma unroll
    for (int ks = 0; ks < 2; ++ks)
      #pragma unroll
      for (int n2 = 0; n2 < 2; ++n2)
        #pragma unroll
        for (int v = 0; v < 8; ++v) {
          int kk = ks*32 + (half<<4) + (v<<1);
          int dc = 512 + h*32 + n2*16 + l16;
          P2 t; t.h[0] = qk[kk*QS + dc]; t.h[1] = qk[(kk+1)*QS + dc];
          bv[ks][n2].u[v] = t.u;
        }
    #pragma unroll 1
    for (int mt = 0; mt < 4; ++mt) {
      ABu a0, a1;
      #pragma unroll
      for (int v = 0; v < 8; ++v) {
        int kb = (half<<4) + (v<<1);
        a0.u[v] = *(const uint32_t*)&Asc[(mt*16 + l16)*AS + kb];
        a1.u[v] = *(const uint32_t*)&Asc[(mt*16 + l16)*AS + 32 + kb];
      }
      #pragma unroll
      for (int n2 = 0; n2 < 2; ++n2) {
        v8f cc = {0.f,0.f,0.f,0.f,0.f,0.f,0.f,0.f};
        cc = wmma16(a0, bv[0][n2], cc);
        cc = wmma16(a1, bv[1][n2], cc);
        #pragma unroll
        for (int r = 0; r < 8; ++r)
          xT[(mt*16 + (half<<3) + r)*XS + h*32 + n2*16 + l16] = (_Float16)cc[r];
      }
    }
  }
  __syncthreads();

  // ---- phase 4: depthwise 5x5 on p-branch, one channel per thread ----
  {
    const int c = tid;
    float w[25];
    #pragma unroll
    for (int i = 0; i < 25; ++i) w[i] = dw_w[c*25 + i];
    float bc = dw_b[c];
    float img[64];
    #pragma unroll
    for (int n = 0; n < 64; ++n) img[n] = (float)qk[n*QS + 768 + c];
    #pragma unroll
    for (int i = 0; i < 8; ++i)
      #pragma unroll
      for (int j = 0; j < 8; ++j) {
        float acc = bc;
        #pragma unroll
        for (int di = 0; di < 5; ++di) {
          int ii = i + di - 2;
          if (ii < 0 || ii > 7) continue;
          #pragma unroll
          for (int dj = 0; dj < 5; ++dj) {
            int jj = j + dj - 2;
            if (jj < 0 || jj > 7) continue;
            acc = fmaf(w[di*5+dj], img[ii*8+jj], acc);
          }
        }
        scr[c*AS + i*8 + j] = (_Float16)acc;   // dwout[channel][pixel]
      }
  }
  __syncthreads();

  // ---- phase 5: grouped 1x1 via WMMA (per head: 32x64 = pw(32x32) @ dwout(32x64)) ----
  {
    const int h = wave;
    #pragma unroll 1
    for (int m2 = 0; m2 < 2; ++m2) {
      float pb[8];
      #pragma unroll
      for (int r = 0; r < 8; ++r) pb[r] = pw_b[h*32 + m2*16 + (half<<3) + r];
      ABu ap;
      #pragma unroll
      for (int v = 0; v < 8; ++v) {
        int kb = (half<<4) + (v<<1);
        ap.u[v] = *(const uint32_t*)&pwW[(h*32 + m2*16 + l16)*32 + kb];
      }
      #pragma unroll
      for (int nt = 0; nt < 4; ++nt) {
        ABu bp;
        #pragma unroll
        for (int v = 0; v < 8; ++v) {
          int kk = (half<<4) + (v<<1);
          int px = nt*16 + l16;
          P2 t; t.h[0] = scr[(h*32 + kk)*AS + px]; t.h[1] = scr[(h*32 + kk + 1)*AS + px];
          bp.u[v] = t.u;
        }
        v8f cc;
        #pragma unroll
        for (int r = 0; r < 8; ++r) cc[r] = pb[r];
        cc = wmma16(ap, bp, cc);
        #pragma unroll
        for (int r = 0; r < 8; ++r) {
          int colc = h*32 + m2*16 + (half<<3) + r;
          int px = nt*16 + l16;
          _Float16* p = &xT[px*XS + colc];
          *p = (_Float16)((float)*p + cc[r]);
        }
      }
    }
  }
  __syncthreads();

  // ---- phase 6: proj GEMM (out64x256 @ proj_w 256x256 + b) -> global f32 ----
  {
    #pragma unroll 1
    for (int mt = 0; mt < 4; ++mt) {
      #pragma unroll 1
      for (int jl = 0; jl < 2; ++jl) {
        int col = wave*32 + jl*16 + l16;
        float bb = proj_b[col];
        v8f acc;
        #pragma unroll
        for (int r = 0; r < 8; ++r) acc[r] = bb;
        #pragma unroll
        for (int ks = 0; ks < 8; ++ks) {
          ABu aO, bW;
          #pragma unroll
          for (int v = 0; v < 8; ++v) {
            int kb = ks*32 + (half<<4) + (v<<1);
            aO.u[v] = *(const uint32_t*)&xT[(mt*16 + l16)*XS + kb];
            int kp = ks*16 + (half<<3) + v;
            bW.u[v] = Wp[kp*256 + col];
          }
          acc = wmma16(aO, bW, acc);
        }
        #pragma unroll
        for (int r = 0; r < 8; ++r) {
          int row = mt*16 + (half<<3) + r;
          out[(blk*64 + (size_t)row)*256 + col] = acc[r];
        }
      }
    }
  }
}

extern "C" void kernel_launch(void* const* d_in, const int* in_sizes, int n_in,
                              void* d_out, int out_size, void* d_ws, size_t ws_size,
                              hipStream_t stream) {
  (void)in_sizes; (void)n_in; (void)out_size; (void)ws_size;
  const float* x      = (const float*)d_in[0];
  const float* qkvp_w = (const float*)d_in[1];
  const float* qkvp_b = (const float*)d_in[2];
  const float* rpb    = (const float*)d_in[3];
  const float* dw_w   = (const float*)d_in[4];
  const float* dw_b   = (const float*)d_in[5];
  const float* pw_w   = (const float*)d_in[6];
  const float* pw_b   = (const float*)d_in[7];
  const float* proj_w = (const float*)d_in[8];
  const float* proj_b = (const float*)d_in[9];
  const int*   rel_idx = (const int*)d_in[10];
  unsigned char* ws = (unsigned char*)d_ws;
  float* out = (float*)d_out;

  remsa_prep<<<800, 256, 0, stream>>>(qkvp_w, proj_w, pw_w, rpb, rel_idx, ws);

  hipFuncSetAttribute((const void*)remsa_main,
                      hipFuncAttributeMaxDynamicSharedMemorySize, SMEM_BYTES);
  remsa_main<<<4096, 256, SMEM_BYTES, stream>>>(x, qkvp_b, dw_w, dw_b, pw_b,
                                                proj_b, ws, out);
}
